// NARM_37409165148968
// MI455X (gfx1250) — matrix operand order repaired and checked by reading, not compiled, not run on hardware
//
#include <hip/hip_runtime.h>
#include <hip/hip_bf16.h>

typedef __bf16 bf16;
typedef __attribute__((ext_vector_type(16))) __bf16 v16bf;
typedef __attribute__((ext_vector_type(8)))  __bf16 v8bf;
typedef __attribute__((ext_vector_type(4)))  __bf16 v4bf;
typedef __attribute__((ext_vector_type(8)))  float  v8f;
typedef __attribute__((ext_vector_type(4)))  float  v4f;

#define T_   200
#define B_   1024
#define I_   128
#define H_   128
#define G3   384
#define LBL_ 4
#define BT   32   // batch rows per persistent GRU block
#define XP   (I_ + 8)   // LDS pitch (bf16 elems) for x/h tiles

static __device__ __forceinline__ float sigf(float x) {
  return 1.0f / (1.0f + __expf(-x));
}

static __device__ __forceinline__ v8f splat8(float v) {
  v8f r;
#pragma unroll
  for (int i = 0; i < 8; ++i) r[i] = v;
  return r;
}

static __device__ __forceinline__ v16bf cat16(v8bf lo, v8bf hi) {
  return __builtin_shufflevector(lo, hi, 0, 1, 2, 3, 4, 5, 6, 7,
                                         8, 9, 10, 11, 12, 13, 14, 15);
}

// A-matrix fragment (16x32 bf16, M x K). Row m = mbase + lane%16.
// lanes 0-15 hold K = {0..7, 16..23}+32*kt ; lanes 16-31 hold K = {8..15, 24..31}+32*kt
static __device__ __forceinline__ v16bf load_frag_a(const bf16* base, int pitch,
                                                    int mbase, int kt, int lane) {
  const int m  = mbase + (lane & 15);
  const int kb = kt * 32 + ((lane & 16) ? 8 : 0);
  const bf16* p = base + (size_t)m * pitch + kb;
  v8bf lo = *(const v8bf*)(p);        // K = kb .. kb+7
  v8bf hi = *(const v8bf*)(p + 16);   // K = kb+16 .. kb+23
  return cat16(lo, hi);
}

// B-matrix fragment (32x16 bf16, K x N) taken from row-major W[n][k] (B = W^T):
// lanes 0-15: col n = nbase+lane%16, K rows = 32*kt + {0..7, 16..23}
// lanes 16-31: same cols, K rows = 32*kt + {8..15, 24..31}
static __device__ __forceinline__ v16bf load_frag_b(const bf16* w, int pitch,
                                                    int nbase, int kt, int lane) {
  const int n = nbase + (lane & 15);
  const int k = kt * 32 + ((lane & 16) ? 8 : 0);
  const bf16* p = w + (size_t)n * pitch + k;
  v8bf lo = *(const v8bf*)(p);
  v8bf hi = *(const v8bf*)(p + 16);
  return cat16(lo, hi);
}

static __device__ __forceinline__ v8f wmma_bf16(v16bf a, v16bf b, v8f c) {
  // D = A(16x32) x B(32x16) + C, f32 accumulate
  return __builtin_amdgcn_wmma_f32_16x16x32_bf16(false, a, false, b, (short)0, c,
                                                 false, false);
}

// CDNA5 async memory->LDS DMA (ASYNCcnt-tracked), GV addressing mode.
static __device__ __forceinline__ void async_copy_b128(unsigned lds_off, const void* g) {
  asm volatile("global_load_async_to_lds_b128 %0, %1, off"
               :: "v"(lds_off), "v"(g)
               : "memory");
}
static __device__ __forceinline__ void wait_asynccnt0() {
  asm volatile("s_wait_asynccnt 0x0" ::: "memory");
}

// ---------------------------------------------------------------------------
// Kernel 0: vectorized f32 -> bf16 conversion (weights + x; all sizes % 4 == 0)
// ---------------------------------------------------------------------------
__global__ void __launch_bounds__(256) cvt4_kernel(const v4f* __restrict__ src,
                                                   v4bf* __restrict__ dst, int n4) {
  int i = blockIdx.x * 256 + threadIdx.x;
  if (i < n4) {
    v4f s = src[i];
    v4bf d;
#pragma unroll
    for (int e = 0; e < 4; ++e) d[e] = (bf16)s[e];
    dst[i] = d;
  }
}

// ---------------------------------------------------------------------------
// Kernel 1: persistent GRU. grid = (B/BT, 2); blockIdx.y selects GRU (g / l).
// Each of 8 waves owns the (r,z,n) gate triple for 16 h-columns; weight
// B-fragments live in VGPRs for all 200 steps. x_{t+1} tile is prefetched
// into a double-buffered LDS tile with async LDS-DMA while step t computes.
// ---------------------------------------------------------------------------
__global__ void __launch_bounds__(256) gru_kernel(
    const bf16*  __restrict__ xbf,    // [T*B, I] bf16
    const bf16*  __restrict__ wbf,    // [wih_g | whh_g | wih_l | whh_l], each [3H][128]
    const float* __restrict__ bih_g, const float* __restrict__ bhh_g,
    const float* __restrict__ bih_l, const float* __restrict__ bhh_l,
    bf16* __restrict__ hg_out, bf16* __restrict__ hl_out) {
  const int gru = blockIdx.y;
  const bf16* wih = wbf + (size_t)gru * 2 * (G3 * I_);
  const bf16* whh = wih + (size_t)G3 * I_;
  const float* bih = gru ? bih_l : bih_g;
  const float* bhh = gru ? bhh_l : bhh_g;
  bf16* hout = gru ? hl_out : hg_out;

  const int b0   = blockIdx.x * BT;
  const int tid  = threadIdx.x;
  const int wave = tid >> 5;
  const int lane = tid & 31;
  const int j    = wave * 16 + (lane & 15);   // h-column owned by this lane

  __shared__ bf16 xt[2][BT][XP];
  __shared__ bf16 ht[BT][XP];

  // LDS byte offsets for async DMA destinations (flat LDS aperture: low 32 bits)
  const unsigned xt_base   = (unsigned)(size_t)&xt[0][0][0];
  const unsigned buf_bytes = BT * XP * 2;

  // Per-thread prefetch slice: 2 x 16B of the 8KB tile (32 rows x 256B)
  const int pf_r0 = tid >> 4;            // rows 0..15
  const int pf_cb = (tid & 15) * 16;     // byte column within row

  // Resident weight fragments: wi/wh[gate][ktile]
  v16bf wi[3][4], wh[3][4];
#pragma unroll
  for (int g = 0; g < 3; ++g)
#pragma unroll
    for (int kt = 0; kt < 4; ++kt) {
      wi[g][kt] = load_frag_b(wih, I_, g * H_ + wave * 16, kt, lane);
      wh[g][kt] = load_frag_b(whh, H_, g * H_ + wave * 16, kt, lane);
    }
  float bi[3], bh[3];
#pragma unroll
  for (int g = 0; g < 3; ++g) { bi[g] = bih[g * H_ + j]; bh[g] = bhh[g * H_ + j]; }

  // h0 = 0
  for (int i = tid; i < BT * H_; i += 256) ht[i >> 7][i & 127] = (bf16)0.0f;

  // Prefetch x tile for t = 0 into buffer 0
  {
    const bf16* src = xbf + ((size_t)0 * B_ + b0) * I_;
    async_copy_b128(xt_base + (unsigned)(pf_r0 * (XP * 2) + pf_cb),
                    (const char*)src + pf_r0 * (I_ * 2) + pf_cb);
    async_copy_b128(xt_base + (unsigned)((pf_r0 + 16) * (XP * 2) + pf_cb),
                    (const char*)src + (pf_r0 + 16) * (I_ * 2) + pf_cb);
  }

  for (int t = 0; t < T_; ++t) {
    const int buf = t & 1;
    wait_asynccnt0();      // this wave's DMA into buf is done
    __syncthreads();       // everyone's DMA done; prev-step ht writes visible

    // Kick off DMA for x_{t+1} into the other buffer (hidden behind WMMAs)
    if (t + 1 < T_) {
      const unsigned db = xt_base + (buf ^ 1) * buf_bytes;
      const bf16* src = xbf + ((size_t)(t + 1) * B_ + b0) * I_;
      async_copy_b128(db + (unsigned)(pf_r0 * (XP * 2) + pf_cb),
                      (const char*)src + pf_r0 * (I_ * 2) + pf_cb);
      async_copy_b128(db + (unsigned)((pf_r0 + 16) * (XP * 2) + pf_cb),
                      (const char*)src + (pf_r0 + 16) * (I_ * 2) + pf_cb);
    }

    // gi = x_t @ Wih^T + bih ; gh = h @ Whh^T + bhh  (kept separate: n-gate needs r*gh_n)
    v8f gi[2][3], gh[2][3];
#pragma unroll
    for (int mt = 0; mt < 2; ++mt)
#pragma unroll
      for (int g = 0; g < 3; ++g) { gi[mt][g] = splat8(bi[g]); gh[mt][g] = splat8(bh[g]); }

    const bf16* xtile = &xt[buf][0][0];
#pragma unroll
    for (int kt = 0; kt < 4; ++kt) {
      v16bf ax0 = load_frag_a(xtile,      XP, 0,  kt, lane);
      v16bf ax1 = load_frag_a(xtile,      XP, 16, kt, lane);
      v16bf ah0 = load_frag_a(&ht[0][0],  XP, 0,  kt, lane);
      v16bf ah1 = load_frag_a(&ht[0][0],  XP, 16, kt, lane);
#pragma unroll
      for (int g = 0; g < 3; ++g) {
        gi[0][g] = wmma_bf16(ax0, wi[g][kt], gi[0][g]);
        gi[1][g] = wmma_bf16(ax1, wi[g][kt], gi[1][g]);
        gh[0][g] = wmma_bf16(ah0, wh[g][kt], gh[0][g]);
        gh[1][g] = wmma_bf16(ah1, wh[g][kt], gh[1][g]);
      }
    }
    __syncthreads();  // all waves done reading ht as A-fragments

    // Gate nonlinearity + h update. C-layout: elem e -> M = e + 8*(lane>=16), N = lane%16.
#pragma unroll
    for (int mt = 0; mt < 2; ++mt) {
#pragma unroll
      for (int e = 0; e < 8; ++e) {
        const int m = mt * 16 + e + ((lane >> 4) << 3);
        float r  = sigf(gi[mt][0][e] + gh[mt][0][e]);
        float z  = sigf(gi[mt][1][e] + gh[mt][1][e]);
        float nn = tanhf(gi[mt][2][e] + r * gh[mt][2][e]);
        float hp = (float)ht[m][j];
        float hn = (1.0f - z) * nn + z * hp;
        ht[m][j] = (bf16)hn;
        hout[((size_t)t * B_ + b0 + m) * H_ + j] = (bf16)hn;
      }
    }
    // next iteration's barrier orders ht writes vs. next GEMM reads
  }
}

// ---------------------------------------------------------------------------
// Kernel 2: scores[tb] = v1 . sigmoid(h_l@A1^T + h_g@A2^T). One 16-row tile
// per wave; 64 WMMAs per wave; width-16 shuffle reduction over N.
// ---------------------------------------------------------------------------
__global__ void __launch_bounds__(256) score_kernel(
    const bf16* __restrict__ hl, const bf16* __restrict__ hg,
    const bf16* __restrict__ a1, const bf16* __restrict__ a2,
    const float* __restrict__ v1, float* __restrict__ scores) {
  const int lane = threadIdx.x & 31;
  const int wave = threadIdx.x >> 5;
  const int tb0  = (blockIdx.x * 8 + wave) * 16;

  v16bf al[4], ag[4];
#pragma unroll
  for (int kt = 0; kt < 4; ++kt) {
    al[kt] = load_frag_a(hl, H_, tb0, kt, lane);
    ag[kt] = load_frag_a(hg, H_, tb0, kt, lane);
  }

  float sc[8];
#pragma unroll
  for (int e = 0; e < 8; ++e) sc[e] = 0.0f;

#pragma unroll
  for (int nt = 0; nt < 8; ++nt) {
    v8f acc = splat8(0.0f);
#pragma unroll
    for (int kt = 0; kt < 4; ++kt) {
      acc = wmma_bf16(al[kt], load_frag_b(a1, H_, nt * 16, kt, lane), acc);
      acc = wmma_bf16(ag[kt], load_frag_b(a2, H_, nt * 16, kt, lane), acc);
    }
    const float vn = v1[nt * 16 + (lane & 15)];
#pragma unroll
    for (int e = 0; e < 8; ++e) sc[e] += vn * sigf(acc[e]);
  }

  // reduce over the 16 lanes sharing the same M (wave32 halves)
#pragma unroll
  for (int e = 0; e < 8; ++e) {
    float v = sc[e];
    v += __shfl_xor(v, 1, 16);
    v += __shfl_xor(v, 2, 16);
    v += __shfl_xor(v, 4, 16);
    v += __shfl_xor(v, 8, 16);
    sc[e] = v;
  }
  if ((lane & 15) == 0) {
    const int half = lane >> 4;
#pragma unroll
    for (int e = 0; e < 8; ++e) scores[tb0 + e + half * 8] = sc[e];
  }
}

// ---------------------------------------------------------------------------
// Kernel 3: out[b,k,:] = cumsum_{s<=idx_k}(score*h_l)[b] + h_g[idx_k,b],
// idx_k = L_b-4+k. One block per batch row, h-parallel, coalesced bf16 reads.
// ---------------------------------------------------------------------------
__global__ void __launch_bounds__(128) out_kernel(
    const bf16* __restrict__ hl, const bf16* __restrict__ hg,
    const float* __restrict__ scores, const int* __restrict__ lengths,
    float* __restrict__ out) {
  const int b = blockIdx.x;
  const int h = threadIdx.x;
  int L = lengths[b];
  if (L < LBL_) L = LBL_;
  if (L > T_)  L = T_;
  const int i0 = L - LBL_;
  float acc = 0.0f;
  for (int s = 0; s < L; ++s) {
    const size_t idx = (size_t)s * B_ + b;
    acc += scores[idx] * (float)hl[idx * H_ + h];
    if (s >= i0)
      out[((size_t)b * LBL_ + (s - i0)) * H_ + h] = acc + (float)hg[idx * H_ + h];
  }
}

// ---------------------------------------------------------------------------
extern "C" void kernel_launch(void* const* d_in, const int* in_sizes, int n_in,
                              void* d_out, int out_size, void* d_ws, size_t ws_size,
                              hipStream_t stream) {
  (void)in_sizes; (void)n_in; (void)out_size; (void)ws_size;
  const float* x      = (const float*)d_in[0];
  const int*   lens   = (const int*)d_in[1];
  const float* Wih_g  = (const float*)d_in[2];
  const float* Whh_g  = (const float*)d_in[3];
  const float* bih_g  = (const float*)d_in[4];
  const float* bhh_g  = (const float*)d_in[5];
  const float* Wih_l  = (const float*)d_in[6];
  const float* Whh_l  = (const float*)d_in[7];
  const float* bih_l  = (const float*)d_in[8];
  const float* bhh_l  = (const float*)d_in[9];
  const float* A1     = (const float*)d_in[10];
  const float* A2     = (const float*)d_in[11];
  const float* v1     = (const float*)d_in[12];
  float* out = (float*)d_out;

  const size_t TB = (size_t)T_ * B_;

  // Workspace layout (~159 MB):
  bf16*  hl      = (bf16*)d_ws;                 // [T*B, H] bf16
  bf16*  hg      = hl + TB * H_;                // [T*B, H] bf16
  float* scores  = (float*)(hg + TB * H_);      // [T*B] f32
  bf16*  xbf     = (bf16*)(scores + TB);        // [T*B, I] bf16
  bf16*  wbf     = xbf + TB * I_;               // bf16 weights
  bf16*  wih_g_b = wbf;
  bf16*  whh_g_b = wih_g_b + G3 * I_;
  bf16*  wih_l_b = whh_g_b + G3 * H_;
  bf16*  whh_l_b = wih_l_b + G3 * I_;
  bf16*  a1_b    = whh_l_b + G3 * H_;
  bf16*  a2_b    = a1_b + H_ * H_;

  const int nW = G3 * I_ / 4;       // 12288 vec4
  const int nA = H_ * H_ / 4;       // 4096 vec4
  const int nX = (int)(TB * I_ / 4);// 6553600 vec4
  cvt4_kernel<<<(nX + 255) / 256, 256, 0, stream>>>((const v4f*)x, (v4bf*)xbf, nX);
  cvt4_kernel<<<(nW + 255) / 256, 256, 0, stream>>>((const v4f*)Wih_g, (v4bf*)wih_g_b, nW);
  cvt4_kernel<<<(nW + 255) / 256, 256, 0, stream>>>((const v4f*)Whh_g, (v4bf*)whh_g_b, nW);
  cvt4_kernel<<<(nW + 255) / 256, 256, 0, stream>>>((const v4f*)Wih_l, (v4bf*)wih_l_b, nW);
  cvt4_kernel<<<(nW + 255) / 256, 256, 0, stream>>>((const v4f*)Whh_l, (v4bf*)whh_l_b, nW);
  cvt4_kernel<<<(nA + 255) / 256, 256, 0, stream>>>((const v4f*)A1, (v4bf*)a1_b, nA);
  cvt4_kernel<<<(nA + 255) / 256, 256, 0, stream>>>((const v4f*)A2, (v4bf*)a2_b, nA);

  gru_kernel<<<dim3(B_ / BT, 2), 256, 0, stream>>>(
      xbf, wbf, bih_g, bhh_g, bih_l, bhh_l, hg, hl);

  score_kernel<<<(int)(TB / 128), 256, 0, stream>>>(hl, hg, a1_b, a2_b, v1, scores);

  out_kernel<<<B_, H_, 0, stream>>>(hl, hg, scores, lens, out);
}
